// CACBlock_29781303231200
// MI455X (gfx1250) — matrix operand order repaired and checked by reading, not compiled, not run on hardware
//
#include <hip/hip_runtime.h>
#include <hip/hip_bf16.h>

// ---- problem constants (B=32, L=2048, D=512) ----
#define BB   32
#define LL   2048
#define DD   512
#define TWO_D 1024
#define NLAGS 32
#define KTOP  5

typedef __attribute__((ext_vector_type(16))) __bf16 v16bf;
typedef __attribute__((ext_vector_type(8)))  float  v8f;

// ---------------------------------------------------------------------------
// 1) per-(b,l) channel mean: xm[b*L+l] = mean_d x[b,l,d]
// ---------------------------------------------------------------------------
__global__ void k_rowmean(const float* __restrict__ x, float* __restrict__ xm) {
    int row = blockIdx.x;                 // 0..B*L-1
    int tid = threadIdx.x;                // 128 threads
    const float4* xr = (const float4*)(x + (size_t)row * DD);
    float4 v = xr[tid];                   // 512/4 = 128 float4
    __shared__ float red[128];
    red[tid] = v.x + v.y + v.z + v.w;
    __syncthreads();
    for (int o = 64; o > 0; o >>= 1) {
        if (tid < o) red[tid] += red[tid + o];
        __syncthreads();
    }
    if (tid == 0) xm[row] = red[0] * (1.0f / DD);
}

// ---------------------------------------------------------------------------
// 2) lag scores: score[b][i] = mean_t xm[b,t]*xm[b,t+lag_i]
//    lag_i = trunc(1 + 167*i/31)  (== np.linspace(1,168,32).astype(int64))
// ---------------------------------------------------------------------------
__device__ __forceinline__ int lag_of(int i) {
    return (int)(1.0 + (167.0 * (double)i) / 31.0);
}

__global__ void k_scores(const float* __restrict__ xm, float* __restrict__ scores) {
    int b = blockIdx.x >> 5;
    int i = blockIdx.x & 31;
    int lag = lag_of(i);
    int n = LL - lag;
    const float* r = xm + b * LL;
    float s = 0.f;
    for (int t = threadIdx.x; t < n; t += blockDim.x) s += r[t] * r[t + lag];
    __shared__ float red[256];
    red[threadIdx.x] = s;
    __syncthreads();
    for (int o = 128; o > 0; o >>= 1) {
        if (threadIdx.x < o) red[threadIdx.x] += red[threadIdx.x + o];
        __syncthreads();
    }
    if (threadIdx.x == 0) scores[b * NLAGS + i] = red[0] / (float)n;
}

// ---------------------------------------------------------------------------
// 3) top-5 per batch -> (weight, lag) pairs; weight = val / (sum(top5)+1e-6)
// ---------------------------------------------------------------------------
__global__ void k_topk(const float* __restrict__ scores,
                       float* __restrict__ topw, int* __restrict__ topl) {
    int b = threadIdx.x;
    if (b >= BB) return;
    float sc[NLAGS];
    for (int i = 0; i < NLAGS; ++i) sc[i] = scores[b * NLAGS + i];
    float vals[KTOP]; int idx[KTOP];
    float denom = 1e-6f;
    for (int k = 0; k < KTOP; ++k) {
        float best = -__builtin_inff(); int bi = 0;
        for (int i = 0; i < NLAGS; ++i)
            if (sc[i] > best) { best = sc[i]; bi = i; }   // first occurrence on ties
        vals[k] = best; idx[k] = bi; sc[bi] = -__builtin_inff();
        denom += best;
    }
    for (int k = 0; k < KTOP; ++k) {
        topw[b * KTOP + k] = vals[k] / denom;
        topl[b * KTOP + k] = lag_of(idx[k]);
    }
}

// ---------------------------------------------------------------------------
// 4) fused feature build: afeat[row][0:512]   = y_conv (15 taps, 3 dilations)
//                         afeat[row][512:1024]= y_auto (5 weighted circular rolls)
//    stored bf16, row-major [B*L][1024]. x (128MB) fits in 192MB L2.
// ---------------------------------------------------------------------------
__global__ void k_features(const float* __restrict__ x,
                           const float* __restrict__ cw0,   // [D][1][3]
                           const float* __restrict__ cw1,   // [D][1][5]
                           const float* __restrict__ cw2,   // [D][1][7]
                           const float* __restrict__ topw, const int* __restrict__ topl,
                           __bf16* __restrict__ afeat) {
    int row = blockIdx.x;                  // b*L + l
    int b = row >> 11;                     // /2048
    int l = row & (LL - 1);
    int d = threadIdx.x * 4;               // 128 threads * 4 channels
    const float* xb = x + (size_t)b * LL * DD;

    float c0 = 0.f, c1 = 0.f, c2 = 0.f, c3 = 0.f;
#pragma unroll
    for (int j = 0; j < 3; ++j) {          // k=3, dil=1
        int t = l + (j - 1);
        if ((unsigned)t < LL) {
            float4 v = *(const float4*)(xb + (size_t)t * DD + d);
            c0 += v.x * cw0[(d + 0) * 3 + j]; c1 += v.y * cw0[(d + 1) * 3 + j];
            c2 += v.z * cw0[(d + 2) * 3 + j]; c3 += v.w * cw0[(d + 3) * 3 + j];
        }
    }
#pragma unroll
    for (int j = 0; j < 5; ++j) {          // k=5, dil=2
        int t = l + (j - 2) * 2;
        if ((unsigned)t < LL) {
            float4 v = *(const float4*)(xb + (size_t)t * DD + d);
            c0 += v.x * cw1[(d + 0) * 5 + j]; c1 += v.y * cw1[(d + 1) * 5 + j];
            c2 += v.z * cw1[(d + 2) * 5 + j]; c3 += v.w * cw1[(d + 3) * 5 + j];
        }
    }
#pragma unroll
    for (int j = 0; j < 7; ++j) {          // k=7, dil=4
        int t = l + (j - 3) * 4;
        if ((unsigned)t < LL) {
            float4 v = *(const float4*)(xb + (size_t)t * DD + d);
            c0 += v.x * cw2[(d + 0) * 7 + j]; c1 += v.y * cw2[(d + 1) * 7 + j];
            c2 += v.z * cw2[(d + 2) * 7 + j]; c3 += v.w * cw2[(d + 3) * 7 + j];
        }
    }

    float a0 = 0.f, a1 = 0.f, a2 = 0.f, a3 = 0.f;
#pragma unroll
    for (int k = 0; k < KTOP; ++k) {       // roll(x, lag)[l] = x[(l-lag) mod L]
        float w = topw[b * KTOP + k];
        int lag = topl[b * KTOP + k];
        int t = l - lag; if (t < 0) t += LL;
        float4 v = *(const float4*)(xb + (size_t)t * DD + d);
        a0 += w * v.x; a1 += w * v.y; a2 += w * v.z; a3 += w * v.w;
    }

    union { __bf16 h[4]; uint2 u; } pk;
    __bf16* base = afeat + (size_t)row * TWO_D;
    pk.h[0] = (__bf16)c0; pk.h[1] = (__bf16)c1; pk.h[2] = (__bf16)c2; pk.h[3] = (__bf16)c3;
    *(uint2*)(base + d) = pk.u;
    pk.h[0] = (__bf16)a0; pk.h[1] = (__bf16)a1; pk.h[2] = (__bf16)a2; pk.h[3] = (__bf16)a3;
    *(uint2*)(base + DD + d) = pk.u;
}

// ---------------------------------------------------------------------------
// 5) cast proj_w ([N=512][K=1024] row-major == B-column layout) to bf16
// ---------------------------------------------------------------------------
__global__ void k_packw(const float* __restrict__ w, __bf16* __restrict__ wb) {
    int i = blockIdx.x * blockDim.x + threadIdx.x;   // 512*1024 elements
    wb[i] = (__bf16)w[i];
}

// ---------------------------------------------------------------------------
// 6) GEMM: out[g,n] = x[g,n] + bias[n] + sum_k afeat[g,k]*wb[n,k]
//    Block tile M=128 x N=128, 8 waves = 4(M) x 2(N), wave tile 32x64
//    (2 A frags x 4 B frags -> 8 WMMAs per K-step of 32).
//    LDS double-buffered: one barrier per K-step; global prefetch for step
//    i+1 issued under the WMMAs of step i.
// ---------------------------------------------------------------------------
#define APAD 40    // 32 K-halfs padded to 40 (80B row, keeps 16B alignment)

__global__ __launch_bounds__(256) void k_gemm(const __bf16* __restrict__ afeat,
                                              const __bf16* __restrict__ wb,
                                              const float*  __restrict__ x,
                                              const float*  __restrict__ bias,
                                              float* __restrict__ out) {
    __shared__ __bf16 At[2][128 * APAD];   // A tile [128 rows][32 k], double buffered
    __shared__ __bf16 Bt[2][128 * APAD];   // B tile [128 cols][32 k] (k contiguous)

    const int mbase = blockIdx.x * 128;
    const int nbase = blockIdx.y * 128;
    const int tid  = threadIdx.x;
    const int lane = tid & 31;
    const int wv   = tid >> 5;
    const int wm   = (wv >> 1) * 32;     // wave M offset within tile (0,32,64,96)
    const int wn   = (wv & 1) * 64;      // wave N offset within tile (0,64)
    const int m    = lane & 15;
    const int kg   = lane >> 4;          // K-group select per ISA layout

    const v8f vzero = {0.f, 0.f, 0.f, 0.f, 0.f, 0.f, 0.f, 0.f};
    v8f acc[2][4] = {{vzero, vzero, vzero, vzero}, {vzero, vzero, vzero, vzero}};

    // LDS fill indices: 128 rows x 2 chunks of 16 halfs (32B) per matrix
    const int fr = tid >> 1;             // 0..127
    const int fc = (tid & 1) * 16;       // 0 or 16 halfs
    const __bf16* aptr = &afeat[(size_t)(mbase + fr) * TWO_D + fc];
    const __bf16* bptr = &wb[(size_t)(nbase + fr) * TWO_D + fc];

    // stage K-step 0
    uint4 pa0 = *(const uint4*)(aptr);
    uint4 pa1 = *(const uint4*)(aptr + 8);
    uint4 pb0 = *(const uint4*)(bptr);
    uint4 pb1 = *(const uint4*)(bptr + 8);

    for (int it = 0; it < TWO_D / 32; ++it) {
        const int buf = it & 1;
        // regs -> LDS for this K-step
        uint4* ad = (uint4*)&At[buf][fr * APAD + fc];
        ad[0] = pa0; ad[1] = pa1;
        uint4* bd = (uint4*)&Bt[buf][fr * APAD + fc];
        bd[0] = pb0; bd[1] = pb1;
        __syncthreads();                 // single barrier per K-step

        // global prefetch of K-step it+1 flies under the WMMAs below
        if (it < TWO_D / 32 - 1) {
            const int k1 = (it + 1) * 32;
            pa0 = *(const uint4*)(aptr + k1);
            pa1 = *(const uint4*)(aptr + k1 + 8);
            pb0 = *(const uint4*)(bptr + k1);
            pb1 = *(const uint4*)(bptr + k1 + 8);
        }

        // A fragments 16x32: lanes<16 hold K {0..7,16..23}, lanes>=16 K {8..15,24..31}
        union { uint4 u[2]; v16bf v; } ua[2];
#pragma unroll
        for (int s = 0; s < 2; ++s) {
            const __bf16* ap = &At[buf][(wm + s * 16 + m) * APAD + kg * 8];
            ua[s].u[0] = *(const uint4*)(ap);
            ua[s].u[1] = *(const uint4*)(ap + 16);
        }
#pragma unroll
        for (int t = 0; t < 4; ++t) {
            // B fragment 32x16: lanes<16 hold K {0..7,16..23}, lanes>=16 K {8..15,24..31} of column n
            union { uint4 u[2]; v16bf v; } ub;
            const __bf16* bp = &Bt[buf][(wn + t * 16 + m) * APAD + kg * 8];
            ub.u[0] = *(const uint4*)(bp);
            ub.u[1] = *(const uint4*)(bp + 16);
#pragma unroll
            for (int s = 0; s < 2; ++s)
                acc[s][t] = __builtin_amdgcn_wmma_f32_16x16x32_bf16(
                    false, ua[s].v, false, ub.v, (short)0, acc[s][t], false, false);
        }
        // no trailing barrier needed: next step writes buf^1, last read two steps ago
    }

    // epilogue: D layout — VGPR r holds (row = r + 8*kg, col = m); fuse bias+residual
#pragma unroll
    for (int t = 0; t < 4; ++t) {
        int col = nbase + wn + t * 16 + m;
        float bv = bias[col];
#pragma unroll
        for (int s = 0; s < 2; ++s) {
#pragma unroll
            for (int r = 0; r < 8; ++r) {
                int grow = mbase + wm + s * 16 + r + 8 * kg;
                size_t o = (size_t)grow * DD + col;
                out[o] = x[o] + acc[s][t][r] + bv;
            }
        }
    }
}

// ---------------------------------------------------------------------------
// 7) LayerNorm over D (population variance), in place on d_out
// ---------------------------------------------------------------------------
__global__ void k_ln(float* __restrict__ h, const float* __restrict__ g,
                     const float* __restrict__ bt) {
    int row = blockIdx.x;
    int tid = threadIdx.x;                 // 128 threads
    float4* hr = (float4*)(h + (size_t)row * DD);
    float4 v = hr[tid];
    __shared__ float rs[128], rq[128];
    rs[tid] = v.x + v.y + v.z + v.w;
    rq[tid] = v.x * v.x + v.y * v.y + v.z * v.z + v.w * v.w;
    __syncthreads();
    for (int o = 64; o > 0; o >>= 1) {
        if (tid < o) { rs[tid] += rs[tid + o]; rq[tid] += rq[tid + o]; }
        __syncthreads();
    }
    float mu  = rs[0] * (1.0f / DD);
    float var = rq[0] * (1.0f / DD) - mu * mu;
    float inv = rsqrtf(var + 1e-5f);
    float4 gg = ((const float4*)g)[tid];
    float4 bb = ((const float4*)bt)[tid];
    v.x = (v.x - mu) * inv * gg.x + bb.x;
    v.y = (v.y - mu) * inv * gg.y + bb.y;
    v.z = (v.z - mu) * inv * gg.z + bb.z;
    v.w = (v.w - mu) * inv * gg.w + bb.w;
    hr[tid] = v;
}

// ---------------------------------------------------------------------------
// launch
// ---------------------------------------------------------------------------
extern "C" void kernel_launch(void* const* d_in, const int* in_sizes, int n_in,
                              void* d_out, int out_size, void* d_ws, size_t ws_size,
                              hipStream_t stream) {
    const float* x     = (const float*)d_in[0];
    const float* cw0   = (const float*)d_in[1];
    const float* cw1   = (const float*)d_in[2];
    const float* cw2   = (const float*)d_in[3];
    const float* projw = (const float*)d_in[4];
    const float* projb = (const float*)d_in[5];
    const float* lng   = (const float*)d_in[6];
    const float* lnb   = (const float*)d_in[7];
    float* out = (float*)d_out;

    // workspace layout (~135.5 MB)
    char* ws = (char*)d_ws;
    __bf16* afeat  = (__bf16*)ws;                                   // 65536*1024*2 = 128 MB
    __bf16* wbf    = (__bf16*)(ws + (size_t)134217728);             // 512*1024*2  = 1 MB
    float*  xm     = (float*)(ws + (size_t)134217728 + 1048576);    // 65536*4 = 256 KB
    float*  scores = xm + BB * LL;                                  // 32*32*4
    float*  topw   = scores + BB * NLAGS;                           // 32*5*4
    int*    topl   = (int*)(topw + BB * KTOP);                      // 32*5*4

    const int rows = BB * LL;                                       // 65536

    k_rowmean <<<rows, 128, 0, stream>>>(x, xm);
    k_scores  <<<BB * NLAGS, 256, 0, stream>>>(xm, scores);
    k_topk    <<<1, 32, 0, stream>>>(scores, topw, topl);
    k_features<<<rows, 128, 0, stream>>>(x, cw0, cw1, cw2, topw, topl, afeat);
    k_packw   <<<(DD * TWO_D) / 256, 256, 0, stream>>>(projw, wbf);
    k_gemm    <<<dim3(rows / 128, DD / 128), 256, 0, stream>>>(afeat, wbf, x, projb, out);
    k_ln      <<<rows, 128, 0, stream>>>(out, lng, lnb);
}
